// HierarchicalVideoEncoder_1047972020254
// MI455X (gfx1250) — compile-verified, matrix operand order checked by reading, not run on hardware
//
#include <hip/hip_runtime.h>
#include <hip/hip_bf16.h>
#include <math.h>

// ---------------------------------------------------------------------------
// HierarchicalVideoEncoder for MI455X (gfx1250, wave32, WMMA).
//
// All dense math runs through one generic f16 WMMA GEMM kernel
// (v_wmma_f32_16x16x32_f16, f32 accumulate) with:
//   * double-buffered LDS tiles filled by GLOBAL_LOAD_ASYNC_TO_LDS_B64
//     (ASYNCcnt-tracked, no VGPR round trip), compute overlaps loads
//   * global_prefetch one tile ahead
//   * fused epilogues (scale / bias / residual / gelu / relu, f32+f16 stores)
// Softmax and LayerNorm are small bandwidth-bound row kernels.
// ---------------------------------------------------------------------------

typedef __attribute__((ext_vector_type(16))) _Float16 v16h;
typedef __attribute__((ext_vector_type(8)))  _Float16 v8h;
typedef __attribute__((ext_vector_type(8)))  float    v8f;

#define D_      768
#define NH_     8
#define HD_     96
#define LAT_    64
#define KC_     588      // 3*14*14 im2col depth
#define KVROWS_ 32768    // B*S*T*256 patch tokens
#define KVLEN_  2048     // T*256 tokens per second
#define BS_     16       // B*S seconds
#define BQ_     1024     // B*S*LAT latent rows total
#define SL_     256      // S*LAT tokens per batch in temporal stage
#define M1_     3072
#define F1_     1536

#define BM 128
#define BN 128
#define BK 32
#define LP 40            // LDS row pitch in halves (80B; keeps 16B-aligned frags)

struct GemmP {
  const _Float16* A; const _Float16* B;
  float* C32; _Float16* C16;
  const float* bias; const float* resid;
  long lda, ldb, ldc, ldr;
  long bsA, hsA, bsB, hsB, bsC, hsC;
  int  M, N, K, nh, opB, act;   // opB: 0 = B[K,N], 1 = B[N,K] (transposed)
  float scale;                  // act: 0 none, 1 gelu(tanh), 2 relu
};

// 16B global->LDS copy via two async b64 ops. The 24-bit instruction offset
// is added to BOTH the LDS address (VDST vgpr) and the global address, so a
// single vgpr pair serves both halves. 8B alignment guaranteed (lda mult of
// 4 halves, LDS chunk offsets mult of 16B).
__device__ __forceinline__ void async_cp16(const _Float16* src, _Float16* dst) {
  unsigned lds = (unsigned)(size_t)dst;                 // low 32b = LDS offset
  unsigned long long ga = (unsigned long long)(size_t)src;
  asm volatile("global_load_async_to_lds_b64 %0, %1, off\n\t"
               "global_load_async_to_lds_b64 %0, %1, off offset:8"
               :: "v"(lds), "v"(ga) : "memory");
}
__device__ __forceinline__ void wait_async_all() {
  asm volatile("s_wait_asynccnt 0x0" ::: "memory");
}

// ---------------------------------------------------------------------------
// Generic tiled WMMA GEMM:  C = act( scale*(A x opB(B)) + bias + resid )
// Block: 256 threads = 8 waves, tile 128x128, k-step 32, 2-stage LDS pipe.
// Wave (wave&1, wave>>1) owns a 64x32 sub-tile: acc[4][2] v8f frags.
// LDS layout: As[m][k], Bs[n][k], pitch 40 halves -> fragment reads are two
// 16B ds loads per operand (frag element e: K=(e>=8?16:0)+hi*8+(e&7)).
// ---------------------------------------------------------------------------
__launch_bounds__(256)
__global__ void gemm_f16_wmma(GemmP p) {
  __shared__ __align__(16) _Float16 As[2][BM * LP];
  __shared__ __align__(16) _Float16 Bs[2][BN * LP];

  const int tid  = threadIdx.x;
  const int z    = blockIdx.z;
  const int bb   = z / p.nh, hh = z % p.nh;
  const _Float16* Ab = p.A + (long)bb * p.bsA + (long)hh * p.hsA;
  const _Float16* Bb = p.B + (long)bb * p.bsB + (long)hh * p.hsB;
  const long cOff = (long)bb * p.bsC + (long)hh * p.hsC;

  const int mBase = blockIdx.x * BM;
  const int nBase = blockIdx.y * BN;

  const int wave = tid >> 5, lane = tid & 31;
  const int mr = lane & 15, hi = lane >> 4;
  const int mw = (wave & 1) * 64;
  const int nw = (wave >> 1) * 32;

  // Stage one 128x32 A tile + 128x32 B tile into LDS buffers (async fast path,
  // guarded scalar fallback on M/N/K edges).
  auto stage = [&](int k0, _Float16* Asb, _Float16* Bsb) {
    const int col = (tid & 3) * 8;
    for (int i = 0; i < 2; ++i) {
      const int row = (tid >> 2) + i * 64;
      const int gm  = mBase + row;
      _Float16* dst = &Asb[row * LP + col];
      const _Float16* src = Ab + (long)gm * p.lda + k0 + col;
      if (gm < p.M && (k0 + col + 8) <= p.K) {
        async_cp16(src, dst);
        if (k0 + 2 * BK <= p.K) __builtin_prefetch(src + BK, 0, 1);
      } else {
        for (int j = 0; j < 8; ++j)
          dst[j] = (gm < p.M && (k0 + col + j) < p.K) ? src[j] : (_Float16)0.f;
      }
    }
    if (p.opB) {               // B is [N,K] row-major: direct copy
      for (int i = 0; i < 2; ++i) {
        const int row = (tid >> 2) + i * 64;
        const int gn  = nBase + row;
        _Float16* dst = &Bsb[row * LP + col];
        const _Float16* src = Bb + (long)gn * p.ldb + k0 + col;
        if (gn < p.N && (k0 + col + 8) <= p.K) {
          async_cp16(src, dst);
          if (k0 + 2 * BK <= p.K) __builtin_prefetch(src + BK, 0, 1);
        } else {
          for (int j = 0; j < 8; ++j)
            dst[j] = (gn < p.N && (k0 + col + j) < p.K) ? src[j] : (_Float16)0.f;
        }
      }
    } else {                   // B is [K,N] row-major: transpose into LDS
      const int kk = tid >> 3;            // 0..31
      const int n0 = (tid & 7) * 16;      // 0..112
      const int gk = k0 + kk;
      const _Float16* src = Bb + (long)gk * p.ldb + nBase + n0;
      _Float16 tmp[16];
      if (gk < p.K && (nBase + n0 + 16) <= p.N) {
        ((uint2*)tmp)[0] = ((const uint2*)src)[0];
        ((uint2*)tmp)[1] = ((const uint2*)src)[1];
        ((uint2*)tmp)[2] = ((const uint2*)src)[2];
        ((uint2*)tmp)[3] = ((const uint2*)src)[3];
      } else {
        for (int j = 0; j < 16; ++j)
          tmp[j] = (gk < p.K && (nBase + n0 + j) < p.N) ? src[j] : (_Float16)0.f;
      }
      for (int j = 0; j < 16; ++j) Bsb[(n0 + j) * LP + kk] = tmp[j];
    }
  };

  v8f acc[4][2];
  const v8f vzero = {0.f,0.f,0.f,0.f,0.f,0.f,0.f,0.f};
  for (int i = 0; i < 4; ++i) for (int j = 0; j < 2; ++j) acc[i][j] = vzero;

  const int nkb = (p.K + BK - 1) / BK;

  stage(0, As[0], Bs[0]);
  wait_async_all();
  __syncthreads();

  for (int kb = 0; kb < nkb; ++kb) {
    const int cur = kb & 1, nxt = cur ^ 1;
    if (kb + 1 < nkb) stage((kb + 1) * BK, As[nxt], Bs[nxt]);  // overlap w/ WMMA

    // ---- batch all fragment loads, then 8 back-to-back WMMAs ----
    union V16 { v16h v; v8h h[2]; };
    V16 bf[2], af[4];
    const _Float16* asb = As[cur];
    const _Float16* bsb = Bs[cur];
#pragma unroll
    for (int ni = 0; ni < 2; ++ni) {
      const _Float16* bp = &bsb[(nw + ni * 16 + mr) * LP + hi * 8];
      bf[ni].h[0] = *(const v8h*)bp;
      bf[ni].h[1] = *(const v8h*)(bp + 16);
    }
#pragma unroll
    for (int mi = 0; mi < 4; ++mi) {
      const _Float16* ap = &asb[(mw + mi * 16 + mr) * LP + hi * 8];
      af[mi].h[0] = *(const v8h*)ap;
      af[mi].h[1] = *(const v8h*)(ap + 16);
    }
#pragma unroll
    for (int mi = 0; mi < 4; ++mi) {
      acc[mi][0] = __builtin_amdgcn_wmma_f32_16x16x32_f16(
          false, af[mi].v, false, bf[0].v, (short)0, acc[mi][0], false, false);
      acc[mi][1] = __builtin_amdgcn_wmma_f32_16x16x32_f16(
          false, af[mi].v, false, bf[1].v, (short)0, acc[mi][1], false, false);
    }

    wait_async_all();    // next tile resident before anyone passes the barrier
    __syncthreads();
  }

  // ---- epilogue: C frag layout col = lane&15, row = r + 8*hi ----
  for (int mi = 0; mi < 4; ++mi)
    for (int ni = 0; ni < 2; ++ni)
      for (int r = 0; r < 8; ++r) {
        const int row = mBase + mw + mi * 16 + hi * 8 + r;
        const int col = nBase + nw + ni * 16 + mr;
        if (row < p.M && col < p.N) {
          float v = acc[mi][ni][r] * p.scale;
          if (p.bias)  v += p.bias[col];
          if (p.resid) v += p.resid[(long)row * p.ldr + col];
          if (p.act == 1) {        // gelu (tanh approx, matches jax default)
            v = 0.5f * v * (1.f + tanhf(0.7978845608028654f * (v + 0.044715f * v * v * v)));
          } else if (p.act == 2) { // relu
            v = fmaxf(v, 0.f);
          }
          const long idx = cOff + (long)row * p.ldc + col;
          if (p.C32) p.C32[idx] = v;
          if (p.C16) p.C16[idx] = (_Float16)v;
        }
      }
}

// ---------------------------------------------------------------------------
__launch_bounds__(256)
__global__ void softmax_rows(const float* S, _Float16* P, int L) {
  __shared__ float red[256];
  const int row = blockIdx.x, tid = threadIdx.x;
  const float* s = S + (long)row * L;
  _Float16*    o = P + (long)row * L;
  float mx = -3.4e38f;
  for (int i = tid; i < L; i += 256) mx = fmaxf(mx, s[i]);
  red[tid] = mx; __syncthreads();
  for (int st = 128; st; st >>= 1) { if (tid < st) red[tid] = fmaxf(red[tid], red[tid + st]); __syncthreads(); }
  mx = red[0]; __syncthreads();
  float sum = 0.f;
  for (int i = tid; i < L; i += 256) sum += __expf(s[i] - mx);
  red[tid] = sum; __syncthreads();
  for (int st = 128; st; st >>= 1) { if (tid < st) red[tid] += red[tid + st]; __syncthreads(); }
  const float inv = 1.f / red[0];
  for (int i = tid; i < L; i += 256) o[i] = (_Float16)(__expf(s[i] - mx) * inv);
}

// LayerNorm over rows of length D; input f32 or f16, optional residual add.
__launch_bounds__(256)
__global__ void layernorm_rows(const float* X32, const _Float16* X16, const float* R,
                               const float* g, const float* b,
                               float* o32, _Float16* o16, int D) {
  __shared__ float red[256];
  const int row = blockIdx.x, tid = threadIdx.x;
  const long base = (long)row * D;
  float sum = 0.f, sq = 0.f;
  for (int i = tid; i < D; i += 256) {
    float x = X32 ? X32[base + i] : (float)X16[base + i];
    if (R) x += R[base + i];
    sum += x; sq += x * x;
  }
  red[tid] = sum; __syncthreads();
  for (int st = 128; st; st >>= 1) { if (tid < st) red[tid] += red[tid + st]; __syncthreads(); }
  const float mean = red[0] / D; __syncthreads();
  red[tid] = sq; __syncthreads();
  for (int st = 128; st; st >>= 1) { if (tid < st) red[tid] += red[tid + st]; __syncthreads(); }
  const float var  = red[0] / D - mean * mean;
  const float rstd = rsqrtf(var + 1e-5f);
  for (int i = tid; i < D; i += 256) {
    float x = X32 ? X32[base + i] : (float)X16[base + i];
    if (R) x += R[base + i];
    const float y = (x - mean) * rstd * g[i] + b[i];
    if (o32) o32[base + i] = y;
    if (o16) o16[base + i] = (_Float16)y;
  }
}

// im2col for the 14x14/s14 patch embed: out[frame*256+patch][c*196+kh*14+kw]
__global__ void im2col_k(const float* frames, _Float16* out, long total) {
  long i = (long)blockIdx.x * blockDim.x + threadIdx.x;
  const long stride = (long)gridDim.x * blockDim.x;
  for (; i < total; i += stride) {
    const long row = i / KC_; const int col = (int)(i % KC_);
    const int frame = (int)(row >> 8), patch = (int)(row & 255);
    const int ph = patch >> 4, pw = patch & 15;
    const int c = col / 196, r = col % 196;
    const int kh = r / 14, kw = r % 14;
    out[i] = (_Float16)frames[(((long)frame * 3 + c) * 224 + (ph * 14 + kh)) * 224
                              + (pw * 14 + kw)];
  }
}

__global__ void f32_to_f16_k(const float* s, _Float16* d, long n) {
  long i = (long)blockIdx.x * blockDim.x + threadIdx.x;
  const long stride = (long)gridDim.x * blockDim.x;
  for (; i < n; i += stride) d[i] = (_Float16)s[i];
}

// ---------------------------------------------------------------------------
extern "C" void kernel_launch(void* const* d_in, const int* in_sizes, int n_in,
                              void* d_out, int out_size, void* d_ws, size_t ws_size,
                              hipStream_t stream) {
  (void)in_sizes; (void)n_in; (void)out_size; (void)ws_size;
  const float* frames =(const float*)d_in[0];
  const float* pe_w  =(const float*)d_in[1];  const float* pe_b  =(const float*)d_in[2];
  const float* latents=(const float*)d_in[3];
  const float* ca_qw =(const float*)d_in[4];  const float* ca_qb =(const float*)d_in[5];
  const float* ca_kw =(const float*)d_in[6];  const float* ca_kb =(const float*)d_in[7];
  const float* ca_vw =(const float*)d_in[8];  const float* ca_vb =(const float*)d_in[9];
  const float* ca_ow =(const float*)d_in[10]; const float* ca_ob =(const float*)d_in[11];
  const float* nq_g  =(const float*)d_in[12]; const float* nq_b  =(const float*)d_in[13];
  const float* nkv_g =(const float*)d_in[14]; const float* nkv_b =(const float*)d_in[15];
  const float* no_g  =(const float*)d_in[16]; const float* no_b  =(const float*)d_in[17];
  const float* m1_w  =(const float*)d_in[18]; const float* m1_b  =(const float*)d_in[19];
  const float* m2_w  =(const float*)d_in[20]; const float* m2_b  =(const float*)d_in[21];
  const float* sa_qw =(const float*)d_in[22]; const float* sa_qb =(const float*)d_in[23];
  const float* sa_kw =(const float*)d_in[24]; const float* sa_kb =(const float*)d_in[25];
  const float* sa_vw =(const float*)d_in[26]; const float* sa_vb =(const float*)d_in[27];
  const float* sa_ow =(const float*)d_in[28]; const float* sa_ob =(const float*)d_in[29];
  const float* f1_w  =(const float*)d_in[30]; const float* f1_b  =(const float*)d_in[31];
  const float* f2_w  =(const float*)d_in[32]; const float* f2_b  =(const float*)d_in[33];
  const float* n1_g  =(const float*)d_in[34]; const float* n1_b  =(const float*)d_in[35];
  const float* n2_g  =(const float*)d_in[36]; const float* n2_b  =(const float*)d_in[37];

  char* base = (char*)d_ws;
  size_t off = 0;
  auto alloc  = [&](size_t bytes)->void* {
    off = (off + 255) & ~(size_t)255; void* p = base + off; off += bytes; return p;
  };
  auto allocH = [&](size_t n)->_Float16* { return (_Float16*)alloc(n * 2); };
  auto allocF = [&](size_t n)->float*    { return (float*)   alloc(n * 4); };

  // ---- persistent f16 weights ----
  _Float16* pe_w16 = allocH((size_t)D_ * KC_);
  _Float16* caq16  = allocH((size_t)D_ * D_);
  _Float16* cak16  = allocH((size_t)D_ * D_);
  _Float16* cav16  = allocH((size_t)D_ * D_);
  _Float16* cao16  = allocH((size_t)D_ * D_);
  _Float16* m1w16  = allocH((size_t)M1_ * D_);
  _Float16* m2w16  = allocH((size_t)D_ * M1_);
  _Float16* saq16  = allocH((size_t)D_ * D_);
  _Float16* sak16  = allocH((size_t)D_ * D_);
  _Float16* sav16  = allocH((size_t)D_ * D_);
  _Float16* sao16  = allocH((size_t)D_ * D_);
  _Float16* f1w16  = allocH((size_t)F1_ * D_);
  _Float16* f2w16  = allocH((size_t)D_ * F1_);

  // ---- persistent activations ----
  _Float16* kv16   = allocH((size_t)KVROWS_ * D_);
  _Float16* kvLN16 = allocH((size_t)KVROWS_ * D_);
  _Float16* Kp16   = allocH((size_t)KVROWS_ * D_);
  _Float16* Vp16   = allocH((size_t)KVROWS_ * D_);
  _Float16* qLN16  = allocH((size_t)LAT_ * D_);
  _Float16* Qp16   = allocH((size_t)LAT_ * D_);
  _Float16* O16    = allocH((size_t)BQ_ * D_);
  float*    out32  = allocF((size_t)BQ_ * D_);
  _Float16* h16    = allocH((size_t)BQ_ * D_);
  _Float16* g16    = allocH((size_t)BQ_ * M1_);
  float*    x32    = allocF((size_t)BQ_ * D_);
  _Float16* x16    = allocH((size_t)BQ_ * D_);
  _Float16* Qs16   = allocH((size_t)BQ_ * D_);
  _Float16* Ks16   = allocH((size_t)BQ_ * D_);
  _Float16* Vs16   = allocH((size_t)BQ_ * D_);
  _Float16* O2_16  = allocH((size_t)BQ_ * D_);
  float*    attn32 = allocF((size_t)BQ_ * D_);
  float*    x2_32  = allocF((size_t)BQ_ * D_);
  _Float16* x2_16  = allocH((size_t)BQ_ * D_);
  _Float16* r16    = allocH((size_t)BQ_ * F1_);
  float*    ff32   = allocF((size_t)BQ_ * D_);

  // ---- reused scratch region (im2col, then attention scores) ----
  off = (off + 255) & ~(size_t)255;
  const size_t scr = off;
  _Float16* Aconv = (_Float16*)(base + scr);                               // 38.5 MB
  float*    Sbuf  = (float*)(base + scr);                                  // 67 MB
  _Float16* P16   = (_Float16*)(base + ((scr + (size_t)BS_*NH_*LAT_*KVLEN_*4 + 255) & ~(size_t)255));
  float*    S2    = (float*)(base + scr);                                  // 8 MB
  _Float16* P2_16 = (_Float16*)(base + ((scr + (size_t)4*NH_*SL_*SL_*4 + 255) & ~(size_t)255));

  auto conv16 = [&](const float* s, _Float16* d, long n) {
    int blocks = (int)((n + 1023) / 1024); if (blocks > 4096) blocks = 4096;
    f32_to_f16_k<<<blocks, 256, 0, stream>>>(s, d, n);
  };
  auto gemm = [&](GemmP p, int Z) {
    dim3 grd((p.M + BM - 1) / BM, (p.N + BN - 1) / BN, Z);
    gemm_f16_wmma<<<grd, 256, 0, stream>>>(p);
  };
  auto GP = [&]() { GemmP p{}; p.nh = 1; p.scale = 1.f; p.opB = 1; return p; };

  // 1) weights -> f16
  conv16(pe_w,  pe_w16, (long)D_*KC_);
  conv16(ca_qw, caq16, (long)D_*D_);  conv16(ca_kw, cak16, (long)D_*D_);
  conv16(ca_vw, cav16, (long)D_*D_);  conv16(ca_ow, cao16, (long)D_*D_);
  conv16(m1_w,  m1w16, (long)M1_*D_); conv16(m2_w,  m2w16, (long)D_*M1_);
  conv16(sa_qw, saq16, (long)D_*D_);  conv16(sa_kw, sak16, (long)D_*D_);
  conv16(sa_vw, sav16, (long)D_*D_);  conv16(sa_ow, sao16, (long)D_*D_);
  conv16(f1_w,  f1w16, (long)F1_*D_); conv16(f2_w,  f2w16, (long)D_*F1_);

  // 2) im2col
  im2col_k<<<16384, 256, 0, stream>>>(frames, Aconv, (long)KVROWS_ * KC_);

  // 3) patch embed GEMM: kv = Aconv @ pe_w^T + pe_b   [32768 x 768]
  { GemmP p = GP(); p.A=Aconv; p.lda=KC_; p.B=pe_w16; p.ldb=KC_; p.bias=pe_b;
    p.C16=kv16; p.ldc=D_; p.M=KVROWS_; p.N=D_; p.K=KC_; gemm(p,1); }

  // 4) LayerNorms for kv and latents
  layernorm_rows<<<KVROWS_,256,0,stream>>>(nullptr, kv16, nullptr, nkv_g, nkv_b, nullptr, kvLN16, D_);
  layernorm_rows<<<LAT_,   256,0,stream>>>(latents, nullptr, nullptr, nq_g, nq_b, nullptr, qLN16, D_);

  // 5) Q/K/V projections (Q identical for all 16 seconds -> compute once)
  { GemmP p = GP(); p.A=qLN16;  p.lda=D_; p.B=caq16; p.ldb=D_; p.bias=ca_qb;
    p.C16=Qp16; p.ldc=D_; p.M=LAT_; p.N=D_; p.K=D_; gemm(p,1); }
  { GemmP p = GP(); p.A=kvLN16; p.lda=D_; p.B=cak16; p.ldb=D_; p.bias=ca_kb;
    p.C16=Kp16; p.ldc=D_; p.M=KVROWS_; p.N=D_; p.K=D_; gemm(p,1); }
  { GemmP p = GP(); p.A=kv16;   p.lda=D_; p.B=cav16; p.ldb=D_; p.bias=ca_vb;
    p.C16=Vp16; p.ldc=D_; p.M=KVROWS_; p.N=D_; p.K=D_; gemm(p,1); }

  // 6) cross-attn scores: S[b,h,q,k] = Qh @ Kh^T / sqrt(96)   Z = 16*8
  { GemmP p = GP(); p.nh=NH_; p.A=Qp16; p.lda=D_; p.bsA=0; p.hsA=HD_;
    p.B=Kp16; p.ldb=D_; p.bsB=(long)KVLEN_*D_; p.hsB=HD_;
    p.C32=Sbuf; p.ldc=KVLEN_; p.bsC=(long)NH_*LAT_*KVLEN_; p.hsC=(long)LAT_*KVLEN_;
    p.M=LAT_; p.N=KVLEN_; p.K=HD_; p.scale=1.f/sqrtf((float)HD_); gemm(p, BS_*NH_); }
  softmax_rows<<<BS_*NH_*LAT_,256,0,stream>>>(Sbuf, P16, KVLEN_);

  // 7) O = P @ V  -> heads concatenated into O16 [1024 x 768]
  { GemmP p = GP(); p.nh=NH_; p.opB=0; p.A=P16; p.lda=KVLEN_;
    p.bsA=(long)NH_*LAT_*KVLEN_; p.hsA=(long)LAT_*KVLEN_;
    p.B=Vp16; p.ldb=D_; p.bsB=(long)KVLEN_*D_; p.hsB=HD_;
    p.C16=O16; p.ldc=D_; p.bsC=(long)LAT_*D_; p.hsC=HD_;
    p.M=LAT_; p.N=HD_; p.K=KVLEN_; gemm(p, BS_*NH_); }

  // 8) out-projection, then Perceiver MLP with residual
  { GemmP p = GP(); p.A=O16; p.lda=D_; p.B=cao16; p.ldb=D_; p.bias=ca_ob;
    p.C32=out32; p.ldc=D_; p.M=BQ_; p.N=D_; p.K=D_; gemm(p,1); }
  layernorm_rows<<<BQ_,256,0,stream>>>(out32, nullptr, nullptr, no_g, no_b, nullptr, h16, D_);
  { GemmP p = GP(); p.A=h16; p.lda=D_; p.B=m1w16; p.ldb=D_; p.bias=m1_b; p.act=1;
    p.C16=g16; p.ldc=M1_; p.M=BQ_; p.N=M1_; p.K=D_; gemm(p,1); }
  { GemmP p = GP(); p.A=g16; p.lda=M1_; p.B=m2w16; p.ldb=M1_; p.bias=m2_b;
    p.resid=out32; p.ldr=D_; p.C32=x32; p.C16=x16; p.ldc=D_;
    p.M=BQ_; p.N=D_; p.K=M1_; gemm(p,1); }   // x = tok, [4*256, 768]

  // 9) temporal self-attention (post-norm encoder layer)
  { GemmP p = GP(); p.A=x16; p.lda=D_; p.B=saq16; p.ldb=D_; p.bias=sa_qb;
    p.C16=Qs16; p.ldc=D_; p.M=BQ_; p.N=D_; p.K=D_; gemm(p,1); }
  { GemmP p = GP(); p.A=x16; p.lda=D_; p.B=sak16; p.ldb=D_; p.bias=sa_kb;
    p.C16=Ks16; p.ldc=D_; p.M=BQ_; p.N=D_; p.K=D_; gemm(p,1); }
  { GemmP p = GP(); p.A=x16; p.lda=D_; p.B=sav16; p.ldb=D_; p.bias=sa_vb;
    p.C16=Vs16; p.ldc=D_; p.M=BQ_; p.N=D_; p.K=D_; gemm(p,1); }
  { GemmP p = GP(); p.nh=NH_; p.A=Qs16; p.lda=D_; p.bsA=(long)SL_*D_; p.hsA=HD_;
    p.B=Ks16; p.ldb=D_; p.bsB=(long)SL_*D_; p.hsB=HD_;
    p.C32=S2; p.ldc=SL_; p.bsC=(long)NH_*SL_*SL_; p.hsC=(long)SL_*SL_;
    p.M=SL_; p.N=SL_; p.K=HD_; p.scale=1.f/sqrtf((float)HD_); gemm(p, 4*NH_); }
  softmax_rows<<<4*NH_*SL_,256,0,stream>>>(S2, P2_16, SL_);
  { GemmP p = GP(); p.nh=NH_; p.opB=0; p.A=P2_16; p.lda=SL_;
    p.bsA=(long)NH_*SL_*SL_; p.hsA=(long)SL_*SL_;
    p.B=Vs16; p.ldb=D_; p.bsB=(long)SL_*D_; p.hsB=HD_;
    p.C16=O2_16; p.ldc=D_; p.bsC=(long)SL_*D_; p.hsC=HD_;
    p.M=SL_; p.N=HD_; p.K=SL_; gemm(p, 4*NH_); }
  { GemmP p = GP(); p.A=O2_16; p.lda=D_; p.B=sao16; p.ldb=D_; p.bias=sa_ob;
    p.C32=attn32; p.ldc=D_; p.M=BQ_; p.N=D_; p.K=D_; gemm(p,1); }
  layernorm_rows<<<BQ_,256,0,stream>>>(x32, nullptr, attn32, n1_g, n1_b, x2_32, x2_16, D_);

  // 10) feed-forward (relu) + final post-norm -> d_out
  { GemmP p = GP(); p.A=x2_16; p.lda=D_; p.B=f1w16; p.ldb=D_; p.bias=f1_b; p.act=2;
    p.C16=r16; p.ldc=F1_; p.M=BQ_; p.N=F1_; p.K=D_; gemm(p,1); }
  { GemmP p = GP(); p.A=r16; p.lda=F1_; p.B=f2w16; p.ldb=F1_; p.bias=f2_b;
    p.C32=ff32; p.ldc=D_; p.M=BQ_; p.N=D_; p.K=F1_; gemm(p,1); }
  layernorm_rows<<<BQ_,256,0,stream>>>(x2_32, nullptr, ff32, n2_g, n2_b, (float*)d_out, nullptr, D_);
}
